// PGNetDecoderModel_25795573579993
// MI455X (gfx1250) — matrix operand order repaired and checked by reading, not compiled, not run on hardware
//
// PGNet pointer-generator decoder for MI455X (gfx1250, wave32, WMMA bf16).
// Roofline: W_proj bf16 (51MB) + enc_feat (33.5MB) are L2-resident (192MB L2);
// HBM traffic ~0.3GB @23.3TB/s -> latency bound, so all GEMMs use
// v_wmma_f32_16x16x32_bf16 with f32 accumulate. W tiles are staged into LDS
// with double-buffered GLOBAL_LOAD_ASYNC_TO_LDS_B128 (ASYNCcnt) when the
// toolchain exposes the builtin; sync b128+ds_store staging otherwise.
#include <hip/hip_runtime.h>
#include <hip/hip_bf16.h>

typedef unsigned short u16;
typedef __attribute__((ext_vector_type(16))) __bf16 v16bf;
typedef __attribute__((ext_vector_type(8)))  float  v8f;

#define B_    32
#define S_    512
#define T_    32
#define EMB_  256
#define HID_  512
#define VOCAB_ 50000
#define OOV_  100
#define VEXT_ 50100

#if defined(__has_builtin)
#if __has_builtin(__builtin_amdgcn_global_load_async_to_lds_b128)
#define HAVE_ASYNC_LDS 1
#endif
#endif
#ifndef HAVE_ASYNC_LDS
#define HAVE_ASYNC_LDS 0
#endif

// ---------------- helpers ----------------
__device__ __forceinline__ float4 ld4(const float* p) {
  return *reinterpret_cast<const float4*>(p);
}
__device__ __forceinline__ u16 bfbits(float f) {
  union { __bf16 b; u16 u; } cv; cv.b = (__bf16)f; return cv.u;
}
__device__ __forceinline__ __bf16 bits2bf(u16 u) {
  union { u16 u_; __bf16 b; } cv; cv.u_ = u; return cv.b;
}
__device__ __forceinline__ float sigf(float x) { return 1.f / (1.f + expf(-x)); }
__device__ __forceinline__ v8f vzero() {
  v8f z;
#pragma unroll
  for (int i = 0; i < 8; ++i) z[i] = 0.f;
  return z;
}
__device__ __forceinline__ void packbf(v16bf& d, float4 p, float4 q, float4 r, float4 s) {
  d[0]=(__bf16)p.x; d[1]=(__bf16)p.y; d[2]=(__bf16)p.z; d[3]=(__bf16)p.w;
  d[4]=(__bf16)q.x; d[5]=(__bf16)q.y; d[6]=(__bf16)q.z; d[7]=(__bf16)q.w;
  d[8]=(__bf16)r.x; d[9]=(__bf16)r.y; d[10]=(__bf16)r.z; d[11]=(__bf16)r.w;
  d[12]=(__bf16)s.x; d[13]=(__bf16)s.y; d[14]=(__bf16)s.z; d[15]=(__bf16)s.w;
}

#if HAVE_ASYNC_LDS
// Builtin parameter types (from hipcc diagnostic): pointer to 16B int vector,
// global (AS1) source and LDS (AS3) destination, plus imm offset + cpol.
typedef int v4i_ __attribute__((vector_size(16)));
typedef __attribute__((address_space(1))) v4i_ gv4i;
typedef __attribute__((address_space(3))) v4i_ lv4i;

// Async copy of 16 bytes per lane, memory -> LDS, tracked by ASYNCcnt.
__device__ __forceinline__ void asyncCopy16(const u16* g, u16* l) {
  __builtin_amdgcn_global_load_async_to_lds_b128(
      (gv4i*)const_cast<u16*>(g), (lv4i*)l, 0, 0);
}
#endif

// ---------------- A-matrix accessors ----------------
struct PlainA {
  const float* A; int lda;
  __device__ __forceinline__ float4 operator()(int m, int k) const {
    return ld4(A + (size_t)m * lda + k);
  }
};
struct EmbCtxA {   // concat(embeddings[ids[:,t]], ctx) : K = 256 + 512
  const float* emb; const int* ids; const float* ctx; int t;
  __device__ __forceinline__ float4 operator()(int m, int k) const {
    if (k < EMB_) {
      int id = ids[m * T_ + t];
      return ld4(emb + (size_t)id * EMB_ + k);
    }
    return ld4(ctx + (size_t)m * HID_ + (k - EMB_));
  }
};
struct Concat2A {  // concat(X, Y), each [32, 512] : K = 1024
  const float* X; const float* Y;
  __device__ __forceinline__ float4 operator()(int m, int k) const {
    return (k < HID_) ? ld4(X + (size_t)m * HID_ + k)
                      : ld4(Y + (size_t)m * HID_ + (k - HID_));
  }
};

// ---------------- wave-level 32x16 WMMA bf16 GEMM tile ----------------
// C(32x16 at n0) += A(32xK, f32 via accessor) * W(KxN, bf16 row-major).
// Per-wave: two 16x16 accumulators; K chunked by 32.
// W tile staged in LDS; async path double-buffers (2 x 32x16 bf16 = 2KB/wave).
template <typename AF>
__device__ __forceinline__ void waveGemmTile(
    const AF& af, const u16* __restrict__ W, int ldw, int K, int n0,
    u16 (*ldsDB)[32][16],          // [2][32][16] per-wave double buffer
    v8f& acc0, v8f& acc1)
{
  const int lane = threadIdx.x & 31;
  const int half = lane >> 4;    // 0 or 1
  const int l15  = lane & 15;

#if HAVE_ASYNC_LDS
  {   // preload chunk 0 into buffer 0
    const u16* wrow = W + (size_t)lane * ldw + n0;
    asyncCopy16(wrow,     &ldsDB[0][lane][0]);
    asyncCopy16(wrow + 8, &ldsDB[0][lane][8]);
  }
#endif

  for (int k0 = 0; k0 < K; k0 += 32) {
#if HAVE_ASYNC_LDS
    const int cur = (k0 >> 5) & 1;
    if (k0 + 32 < K) {           // issue next chunk into the other buffer
      const u16* wrow = W + (size_t)(k0 + 32 + lane) * ldw + n0;
      asyncCopy16(wrow,     &ldsDB[cur ^ 1][lane][0]);
      asyncCopy16(wrow + 8, &ldsDB[cur ^ 1][lane][8]);
      // in-order completion: <=2 outstanding means the current chunk landed
      asm volatile("s_wait_asynccnt 2" ::: "memory");
    } else {
      asm volatile("s_wait_asynccnt 0" ::: "memory");
    }
    const u16 (*ldsW)[16] = ldsDB[cur];
#else
    // Sync staging: each lane copies one 16-col bf16 row (32B) via VGPRs.
    const u16* wrow = W + (size_t)(k0 + lane) * ldw + n0;
    const uint4* wsrc = reinterpret_cast<const uint4*>(wrow);
    uint4* wdst = reinterpret_cast<uint4*>(&ldsDB[0][lane][0]);
    wdst[0] = wsrc[0];
    wdst[1] = wsrc[1];
    if (k0 + 32 < K)
      __builtin_prefetch(W + (size_t)(k0 + 32 + lane) * ldw + n0, 0, 1);
    asm volatile("s_wait_dscnt 0" ::: "memory");
    const u16 (*ldsW)[16] = ldsDB[0];
#endif

    // A fragments (ISA 16-bit A 16x32 layout: lane-half picks K sub-octet).
    const int kb = k0 + (half << 3);
    float4 a0 = af(l15, kb),        a1 = af(l15, kb + 4);
    float4 a2 = af(l15, kb + 16),   a3 = af(l15, kb + 20);
    float4 b0 = af(16 + l15, kb),      b1 = af(16 + l15, kb + 4);
    float4 b2 = af(16 + l15, kb + 16), b3 = af(16 + l15, kb + 20);
    v16bf Af0, Af1, Bf;
    packbf(Af0, a0, a1, a2, a3);
    packbf(Af1, b0, b1, b2, b3);

    // B fragment: lane-half picks K sub-octet (matches A), column = lane%16.
    const int krow = half << 3;
#pragma unroll
    for (int e = 0; e < 16; ++e) Bf[e] = bits2bf(ldsW[krow + e + ((e >> 3) << 3)][l15]);

    acc0 = __builtin_amdgcn_wmma_f32_16x16x32_bf16(false, Af0, false, Bf,
                                                   (short)0, acc0, false, false);
    acc1 = __builtin_amdgcn_wmma_f32_16x16x32_bf16(false, Af1, false, Bf,
                                                   (short)0, acc1, false, false);
  }
}

__device__ __forceinline__ void storeTile(float* __restrict__ C, int ldc,
                                          int m0, int n0,
                                          const v8f& acc0, const v8f& acc1,
                                          const float* bias)
{
  const int lane = threadIdx.x & 31;
  const int l15  = lane & 15;
  const int mrow = (lane >> 4) * 8;
  const float bv = bias ? bias[n0 + l15] : 0.f;
#pragma unroll
  for (int r = 0; r < 8; ++r) {
    C[(size_t)(m0 + mrow + r)      * ldc + n0 + l15] = acc0[r] + bv;
    C[(size_t)(m0 + 16 + mrow + r) * ldc + n0 + l15] = acc1[r] + bv;
  }
}

// ---------------- kernels ----------------
__global__ void k_f32_to_bf16(const float* __restrict__ src, u16* __restrict__ dst, long n) {
  long i = (long)blockIdx.x * blockDim.x + threadIdx.x;
  if (i < n) dst[i] = bfbits(src[i]);
}

// enc_feat[B*S,512] = encoder_states @ W_enc + b_enc   (16384 wave tiles)
__global__ void k_encfeat(const float* __restrict__ encs, const u16* __restrict__ Wenc,
                          const float* __restrict__ benc, float* __restrict__ ef) {
  __shared__ __align__(16) u16 lds[4][2][32][16];
  const int w = threadIdx.x >> 5;
  const int tile = blockIdx.x * 4 + w;       // 0..16383
  const int mtile = tile >> 5, ntile = tile & 31;
  v8f a0 = vzero(), a1 = vzero();
  PlainA af{encs + (size_t)mtile * 32 * HID_, HID_};
  waveGemmTile(af, Wenc, HID_, HID_, ntile * 16, lds[w], a0, a1);
  storeTile(ef + (size_t)mtile * 32 * HID_, HID_, 0, ntile * 16, a0, a1, benc);
}

// x = [emb|ctx]@W_x ; z = x@Wk + h@Wr + b ; gates -> h,c   (single block)
__global__ void k_step1(const float* __restrict__ emb, const int* __restrict__ ids, int t,
                        const float* __restrict__ ctx,
                        const float* __restrict__ hPrev, const float* __restrict__ cPrev,
                        const u16* __restrict__ Wx, const u16* __restrict__ Wk,
                        const u16* __restrict__ Wr, const float* __restrict__ bl,
                        float* __restrict__ x, float* __restrict__ z,
                        float* __restrict__ h, float* __restrict__ c) {
  __shared__ __align__(16) u16 lds[8][2][32][16];
  const int w = threadIdx.x >> 5;
  EmbCtxA aEmb{emb, ids, ctx, t};
  for (int tile = w; tile < 32; tile += 8) {           // x: N=512
    v8f a0 = vzero(), a1 = vzero();
    waveGemmTile(aEmb, Wx, HID_, EMB_ + HID_, tile * 16, lds[w], a0, a1);
    storeTile(x, HID_, 0, tile * 16, a0, a1, nullptr);
  }
  __syncthreads();
  PlainA ax{x, HID_}, ah{hPrev, HID_};
  for (int tile = w; tile < 128; tile += 8) {          // z: N=2048
    v8f a0 = vzero(), a1 = vzero();
    waveGemmTile(ax, Wk, 4 * HID_, HID_, tile * 16, lds[w], a0, a1);
    waveGemmTile(ah, Wr, 4 * HID_, HID_, tile * 16, lds[w], a0, a1);
    storeTile(z, 4 * HID_, 0, tile * 16, a0, a1, bl);
  }
  __syncthreads();
  for (int idx = threadIdx.x; idx < B_ * HID_; idx += blockDim.x) {
    const int b = idx >> 9, j = idx & 511;
    const float* zb = z + (size_t)b * 4 * HID_;
    float ig = zb[j], fg = zb[HID_ + j], gg = zb[2 * HID_ + j], og = zb[3 * HID_ + j];
    float c2 = sigf(fg) * cPrev[idx] + sigf(ig) * tanhf(gg);
    c[idx] = c2;
    h[idx] = sigf(og) * tanhf(c2);
  }
}

// C[32,512] = [X|Y] @ W (K=1024); dec_feat (h|c @ W_attn) and out (h|ctx @ W_out)
__global__ void k_concat_gemm(const float* __restrict__ X, const float* __restrict__ Y,
                              const u16* __restrict__ W, float* __restrict__ C) {
  __shared__ __align__(16) u16 lds[8][2][32][16];
  const int w = threadIdx.x >> 5;
  Concat2A af{X, Y};
  for (int tile = w; tile < 32; tile += 8) {
    v8f a0 = vzero(), a1 = vzero();
    waveGemmTile(af, W, HID_, 2 * HID_, tile * 16, lds[w], a0, a1);
    storeTile(C, HID_, 0, tile * 16, a0, a1, nullptr);
  }
}

// attention: e -> masked softmax -> attn, cov += attn, ctx = attn@enc_feat
__global__ void k_attend(const float* __restrict__ encf, const float* __restrict__ decf,
                         float* __restrict__ cov, const float* __restrict__ v,
                         const float* __restrict__ wcov, const float* __restrict__ bcov,
                         const float* __restrict__ mask, int use_cov,
                         float* __restrict__ attn, float* __restrict__ ctx) {
  __shared__ float df[HID_], vs[HID_], wc[HID_], bc[HID_];
  __shared__ float e_sh[S_], a_sh[S_], red[S_];
  const int b = blockIdx.x, tid = threadIdx.x;
  df[tid] = decf[b * HID_ + tid];
  vs[tid] = v[tid]; wc[tid] = wcov[tid]; bc[tid] = bcov[tid];
  __syncthreads();
  const int w = tid >> 5, lane = tid & 31;
  const float* efb = encf + (size_t)b * S_ * HID_;
  for (int i = 0; i < 32; ++i) {
    const int s = w * 32 + i;
    const float cv = use_cov ? cov[b * S_ + s] : 0.f;
    const float* row = efb + (size_t)s * HID_;
    float p = 0.f;
#pragma unroll 4
    for (int j = 0; j < 16; ++j) {
      const int hh = lane + (j << 5);
      float f = row[hh] + df[hh];
      if (use_cov) f += cv * wc[hh] + bc[hh];
      p += vs[hh] * tanhf(f);
    }
#pragma unroll
    for (int o = 16; o; o >>= 1) p += __shfl_xor(p, o, 32);
    if (!lane) e_sh[s] = p;
  }
  __syncthreads();
  red[tid] = e_sh[tid]; __syncthreads();
  for (int o = 256; o; o >>= 1) { if (tid < o) red[tid] = fmaxf(red[tid], red[tid + o]); __syncthreads(); }
  const float mx = red[0]; __syncthreads();
  const float pe = expf(e_sh[tid] - mx) * mask[b * S_ + tid];
  red[tid] = pe; __syncthreads();
  for (int o = 256; o; o >>= 1) { if (tid < o) red[tid] += red[tid + o]; __syncthreads(); }
  const float a = pe / red[0];
  a_sh[tid] = a;
  attn[b * S_ + tid] = a;
  cov[b * S_ + tid] += a;           // cov zeroed at launch; step t uses sum_{<t}
  __syncthreads();
  float acc = 0.f;
  for (int s = 0; s < S_; ++s) acc += a_sh[s] * efb[(size_t)s * HID_ + tid];
  ctx[b * HID_ + tid] = acc;
}

__global__ void k_pgen(const float* __restrict__ ctx, const float* __restrict__ h,
                       const float* __restrict__ c, const float* __restrict__ x,
                       const float* __restrict__ Wp, float* __restrict__ pg) {
  const int b = threadIdx.x;
  if (b < B_) {
    float a = 0.f;
    for (int k = 0; k < HID_; ++k) {
      a += ctx[b * HID_ + k] * Wp[k]
         + h[b * HID_ + k]   * Wp[HID_ + k]
         + c[b * HID_ + k]   * Wp[2 * HID_ + k]
         + x[b * HID_ + k]   * Wp[3 * HID_ + k];
    }
    pg[b] = sigf(a);
  }
}

// logits[32,50000] = out @ W_proj(bf16) + b_proj   (3125 wave tiles)
__global__ void k_logits(const float* __restrict__ outS, const u16* __restrict__ Wp,
                         const float* __restrict__ bp, float* __restrict__ lg) {
  __shared__ __align__(16) u16 lds[4][2][32][16];
  const int w = threadIdx.x >> 5;
  int tile = blockIdx.x * 4 + w;
  const int NT = VOCAB_ / 16;
  if (tile >= NT) tile = NT - 1;   // clamp (duplicate work; identical values)
  v8f a0 = vzero(), a1 = vzero();
  PlainA af{outS, HID_};
  waveGemmTile(af, Wp, VOCAB_, HID_, tile * 16, lds[w], a0, a1);
  storeTile(lg, VOCAB_, 0, tile * 16, a0, a1, bp);
}

__global__ void k_rowstats(const float* __restrict__ lg, float* __restrict__ rmx,
                           float* __restrict__ rsi) {
  __shared__ float red[256];
  const int b = blockIdx.x, tid = threadIdx.x;
  const float* L = lg + (size_t)b * VOCAB_;
  float m = -3.4e38f;
  for (int j = tid; j < VOCAB_; j += 256) m = fmaxf(m, L[j]);
  red[tid] = m; __syncthreads();
  for (int o = 128; o; o >>= 1) { if (tid < o) red[tid] = fmaxf(red[tid], red[tid + o]); __syncthreads(); }
  const float mx = red[0]; __syncthreads();
  float s = 0.f;
  for (int j = tid; j < VOCAB_; j += 256) s += expf(L[j] - mx);
  red[tid] = s; __syncthreads();
  for (int o = 128; o; o >>= 1) { if (tid < o) red[tid] += red[tid + o]; __syncthreads(); }
  if (!tid) { rmx[b] = mx; rsi[b] = 1.f / red[0]; }
}

__global__ void k_final(const float* __restrict__ lg, const float* __restrict__ rmx,
                        const float* __restrict__ rsi, const float* __restrict__ pg,
                        float* __restrict__ outT) {
  long i = (long)blockIdx.x * blockDim.x + threadIdx.x;
  if (i >= (long)B_ * VEXT_) return;
  const int b = (int)(i / VEXT_);
  const int j = (int)(i - (long)b * VEXT_);
  float val = 0.f;
  if (j < VOCAB_)
    val = pg[b] * expf(lg[(size_t)b * VOCAB_ + j] - rmx[b]) * rsi[b];
  outT[i] = val;
}

__global__ void k_scatter(const int* __restrict__ ids, const float* __restrict__ attn,
                          const float* __restrict__ pg, float* __restrict__ outT) {
  const int i = blockIdx.x * blockDim.x + threadIdx.x;
  if (i >= B_ * S_) return;
  const int b = i >> 9;
  atomicAdd(outT + (size_t)b * VEXT_ + ids[i], (1.f - pg[b]) * attn[i]);
}

// ---------------- workspace layout ----------------
static constexpr size_t AL(size_t x) { return (x + 255) & ~(size_t)255; }
constexpr size_t OFF_EF  = 0;
constexpr size_t OFF_WP  = AL(OFF_EF  + (size_t)B_ * S_ * HID_ * 4);     // enc_feat f32
constexpr size_t OFF_LG  = AL(OFF_WP  + (size_t)HID_ * VOCAB_ * 2);      // W_proj bf16
constexpr size_t OFF_WK  = AL(OFF_LG  + (size_t)B_ * VOCAB_ * 4);        // logits f32
constexpr size_t OFF_WR  = AL(OFF_WK  + (size_t)HID_ * 4 * HID_ * 2);
constexpr size_t OFF_WX  = AL(OFF_WR  + (size_t)HID_ * 4 * HID_ * 2);
constexpr size_t OFF_WA  = AL(OFF_WX  + (size_t)(EMB_ + HID_) * HID_ * 2);
constexpr size_t OFF_WO  = AL(OFF_WA  + (size_t)2 * HID_ * HID_ * 2);
constexpr size_t OFF_WE  = AL(OFF_WO  + (size_t)2 * HID_ * HID_ * 2);
constexpr size_t OFF_H   = AL(OFF_WE  + (size_t)HID_ * HID_ * 2);
constexpr size_t OFF_C   = AL(OFF_H   + (size_t)B_ * HID_ * 4);
constexpr size_t OFF_X   = AL(OFF_C   + (size_t)B_ * HID_ * 4);
constexpr size_t OFF_Z   = AL(OFF_X   + (size_t)B_ * HID_ * 4);
constexpr size_t OFF_CTX = AL(OFF_Z   + (size_t)B_ * 4 * HID_ * 4);
constexpr size_t OFF_ATT = AL(OFF_CTX + (size_t)B_ * HID_ * 4);
constexpr size_t OFF_COV = AL(OFF_ATT + (size_t)B_ * S_ * 4);
constexpr size_t OFF_DF  = AL(OFF_COV + (size_t)B_ * S_ * 4);
constexpr size_t OFF_OUT = AL(OFF_DF  + (size_t)B_ * HID_ * 4);
constexpr size_t OFF_PG  = AL(OFF_OUT + (size_t)B_ * HID_ * 4);
constexpr size_t OFF_RMX = AL(OFF_PG  + (size_t)B_ * 4);
constexpr size_t OFF_RSI = AL(OFF_RMX + (size_t)B_ * 4);

extern "C" void kernel_launch(void* const* d_in, const int* in_sizes, int n_in,
                              void* d_out, int out_size, void* d_ws, size_t ws_size,
                              hipStream_t stream) {
  (void)in_sizes; (void)n_in; (void)out_size; (void)ws_size;
  const int*   dec_ids = (const int*)  d_in[0];
  const int*   enc_ids = (const int*)  d_in[1];
  const float* encs    = (const float*)d_in[2];
  const float* mask    = (const float*)d_in[3];
  const float* init_h  = (const float*)d_in[4];
  const float* init_c  = (const float*)d_in[5];
  const float* embt    = (const float*)d_in[6];
  const float* W_enc   = (const float*)d_in[7];
  const float* b_enc   = (const float*)d_in[8];
  const float* W_attn  = (const float*)d_in[9];
  const float* vv      = (const float*)d_in[10];
  const float* w_cov   = (const float*)d_in[11];
  const float* b_cov   = (const float*)d_in[12];
  const float* W_x     = (const float*)d_in[13];
  const float* W_out   = (const float*)d_in[14];
  const float* Wk      = (const float*)d_in[15];
  const float* Wr      = (const float*)d_in[16];
  const float* b_lstm  = (const float*)d_in[17];
  const float* W_pgen  = (const float*)d_in[18];
  const float* W_proj  = (const float*)d_in[19];
  const float* b_proj  = (const float*)d_in[20];

  char* ws = (char*)d_ws;
  float* ef   = (float*)(ws + OFF_EF);
  u16*   wpB  = (u16*)  (ws + OFF_WP);
  float* lg   = (float*)(ws + OFF_LG);
  u16*   wkB  = (u16*)  (ws + OFF_WK);
  u16*   wrB  = (u16*)  (ws + OFF_WR);
  u16*   wxB  = (u16*)  (ws + OFF_WX);
  u16*   waB  = (u16*)  (ws + OFF_WA);
  u16*   woB  = (u16*)  (ws + OFF_WO);
  u16*   weB  = (u16*)  (ws + OFF_WE);
  float* h    = (float*)(ws + OFF_H);
  float* c    = (float*)(ws + OFF_C);
  float* x    = (float*)(ws + OFF_X);
  float* z    = (float*)(ws + OFF_Z);
  float* ctx  = (float*)(ws + OFF_CTX);
  float* attn = (float*)(ws + OFF_ATT);
  float* cov  = (float*)(ws + OFF_COV);
  float* df   = (float*)(ws + OFF_DF);
  float* outS = (float*)(ws + OFF_OUT);
  float* pg   = (float*)(ws + OFF_PG);
  float* rmx  = (float*)(ws + OFF_RMX);
  float* rsi  = (float*)(ws + OFF_RSI);

  auto conv = [&](const float* s, u16* d, long n) {
    k_f32_to_bf16<<<(unsigned)((n + 255) / 256), 256, 0, stream>>>(s, d, n);
  };
  conv(W_proj, wpB, (long)HID_ * VOCAB_);
  conv(Wk,     wkB, (long)HID_ * 4 * HID_);
  conv(Wr,     wrB, (long)HID_ * 4 * HID_);
  conv(W_x,    wxB, (long)(EMB_ + HID_) * HID_);
  conv(W_attn, waB, (long)2 * HID_ * HID_);
  conv(W_out,  woB, (long)2 * HID_ * HID_);
  conv(W_enc,  weB, (long)HID_ * HID_);

  (void)hipMemsetAsync(ctx, 0, (size_t)B_ * HID_ * 4, stream);
  (void)hipMemsetAsync(cov, 0, (size_t)B_ * S_ * 4, stream);

  k_encfeat<<<4096, 128, 0, stream>>>(encs, weB, b_enc, ef);

  const int nLogitsBlk = (VOCAB_ / 16 + 3) / 4;                 // 782
  const long nFinal = (long)B_ * VEXT_;
  for (int t = 0; t < T_; ++t) {
    const float* hP = t ? h : init_h;
    const float* cP = t ? c : init_c;
    k_step1<<<1, 256, 0, stream>>>(embt, dec_ids, t, ctx, hP, cP,
                                   wxB, wkB, wrB, b_lstm, x, z, h, c);
    k_concat_gemm<<<1, 256, 0, stream>>>(h, c, waB, df);
    k_attend<<<B_, 512, 0, stream>>>(ef, df, cov, vv, w_cov, b_cov, mask,
                                     (t > 0) ? 1 : 0, attn, ctx);
    k_concat_gemm<<<1, 256, 0, stream>>>(h, ctx, woB, outS);
    k_pgen<<<1, 32, 0, stream>>>(ctx, h, c, x, W_pgen, pg);
    k_logits<<<nLogitsBlk, 128, 0, stream>>>(outS, wpB, b_proj, lg);
    k_rowstats<<<B_, 256, 0, stream>>>(lg, rmx, rsi);
    float* outT = (float*)d_out + (size_t)t * B_ * VEXT_;
    k_final<<<(unsigned)((nFinal + 255) / 256), 256, 0, stream>>>(lg, rmx, rsi, pg, outT);
    k_scatter<<<(B_ * S_ + 255) / 256, 256, 0, stream>>>(enc_ids, attn, pg, outT);
  }
}
